// Attention_11828339933743
// MI455X (gfx1250) — compile-verified, matrix operand order checked by reading, not run on hardware
//
#include <hip/hip_runtime.h>
#include <hip/hip_bf16.h>
#include <stdint.h>

typedef __attribute__((ext_vector_type(16))) __bf16        v16bf;
typedef __attribute__((ext_vector_type(8)))  float         v8f;
typedef __attribute__((ext_vector_type(4)))  unsigned int  v4u;
typedef __attribute__((ext_vector_type(4)))  int           v4i;
typedef __attribute__((ext_vector_type(4)))  float         v4f;

struct FragRaw { v4u lo, hi; };            // 32 bytes == v16bf
union  Cvt8    { __bf16 h[8]; v4u u; };    // 8 bf16 <-> 16B

#define DM   1024   // D_MODEL = D_KEY = D_VALUE
#define SEQ  2048
#define NB   4

// ---------------- gfx1250 async global->LDS helpers ----------------
__device__ __forceinline__ void async_ld_b128(const void* g, void* l)
{
#if __has_builtin(__builtin_amdgcn_global_load_async_to_lds_b128)
  __builtin_amdgcn_global_load_async_to_lds_b128(
      (__attribute__((address_space(1))) v4i*)(uintptr_t)g,
      (__attribute__((address_space(3))) v4i*)l, 0, 0);
#else
  unsigned loff = (unsigned)(uintptr_t)(__attribute__((address_space(3))) void*)l;
  unsigned long long ga = (unsigned long long)(uintptr_t)g;
  asm volatile("global_load_async_to_lds_b128 %0, %1, off" :: "v"(loff), "v"(ga) : "memory");
#endif
}

__device__ __forceinline__ void wait_async0()
{
#if __has_builtin(__builtin_amdgcn_s_wait_asynccnt)
  __builtin_amdgcn_s_wait_asynccnt(0);
#else
  asm volatile("s_wait_asynccnt 0x0" ::: "memory");
#endif
}

// ---------------- GEMM config ----------------
constexpr int TM = 128, TN = 128, KC = 32;
constexpr int LROW = 40;   // padded LDS row stride (bf16 elems): 80B rows -> conflict-free b128 frag loads

// D(M x N) = A(M x K) * BT(N x K)^T  (+bias)(*scale)(+mask), batched over blockIdx.z
template<bool HAS_BIAS, bool HAS_MASK, bool OUT_BF16, bool OUT_TRANS, bool A_FP32>
__global__ __launch_bounds__(256)
void gemm_wmma_bf16(const void* __restrict__ Av, const __bf16* __restrict__ Bt,
                    const float* __restrict__ biasAll, const float* __restrict__ maskAll,
                    void* __restrict__ Cv,
                    int M, int N, int K,
                    int lda, int ldb, int ldc,
                    long long strideA, long long strideB, long long strideC, long long strideMask,
                    float scale)
{
  __shared__ __bf16 sA[2][TM * LROW];
  __shared__ __bf16 sB[2][TN * LROW];

  const int tid  = threadIdx.x;
  const int wave = tid >> 5, lane = tid & 31;
  const int lh = lane >> 4, lr = lane & 15;     // lane-half, lane-row
  const int wm = wave & 1,  wn = wave >> 1;     // 2x4 wave grid -> 64x32 per wave
  const int bm = blockIdx.y * TM, bn = blockIdx.x * TN;
  const int b  = blockIdx.z;

  const float*  Af = (const float*) Av + (A_FP32 ? (long long)b * strideA : 0);
  const __bf16* Ab = (const __bf16*)Av + (A_FP32 ? 0 : (long long)b * strideA);
  const __bf16* Bb = Bt + (long long)b * strideB;
  const float*  maskp = HAS_MASK ? (maskAll + (long long)b * strideMask) : nullptr;
  float*  Cf = (float*) Cv + (long long)b * strideC;
  __bf16* Cb = (__bf16*)Cv + (long long)b * strideC;

  // Per-thread staging coordinates: thread covers (row, 8-elem chunk) x 2 iters.
  // Async path: fire-and-forget DMA into LDS (ASYNCcnt), no VGPR round trip.
  auto issueAsync = [&](int buf, int k0) {
#pragma unroll
    for (int it = 0; it < 2; ++it) {
      const int idx = tid + it * 256;   // 0..511
      const int row = idx >> 2;         // 0..127
      const int ch  = idx & 3;          // 8 bf16 elems each
      if constexpr (!A_FP32)
        async_ld_b128(Ab + (long long)(bm + row) * lda + k0 + ch * 8,
                      &sA[buf][row * LROW + ch * 8]);
      async_ld_b128(Bb + (long long)(bn + row) * ldb + k0 + ch * 8,
                    &sB[buf][row * LROW + ch * 8]);
    }
  };

  // fp32-A path: global->VGPR early, convert+LDS-store late (hides load latency behind WMMA)
  v4f ar[2][2];
  auto loadA = [&](int k0) {
    if constexpr (A_FP32) {
#pragma unroll
      for (int it = 0; it < 2; ++it) {
        const int idx = tid + it * 256;
        const float* s = Af + (long long)(bm + (idx >> 2)) * lda + k0 + (idx & 3) * 8;
        ar[it][0] = *(const v4f*)s;
        ar[it][1] = *(const v4f*)(s + 4);
      }
    }
  };
  auto storeA = [&](int buf) {
    if constexpr (A_FP32) {
#pragma unroll
      for (int it = 0; it < 2; ++it) {
        const int idx = tid + it * 256;
        Cvt8 c;
#pragma unroll
        for (int i = 0; i < 4; ++i) {
          c.h[i]     = (__bf16)ar[it][0][i];
          c.h[4 + i] = (__bf16)ar[it][1][i];
        }
        *(v4u*)&sA[buf][(idx >> 2) * LROW + (idx & 3) * 8] = c.u;
      }
    }
  };

  v8f acc[4][2] = {};

  issueAsync(0, 0);
  loadA(0);
  storeA(0);
  wait_async0();
  __syncthreads();

  const int nk = K / KC;
  for (int kb = 0; kb < nk; ++kb) {
    const int cur  = kb & 1;
    const bool more = (kb + 1 < nk);
    if (more) {
      issueAsync(cur ^ 1, (kb + 1) * KC);
      loadA((kb + 1) * KC);
    }

    v16bf afr[4], bfr[2];
#pragma unroll
    for (int mt = 0; mt < 4; ++mt) {
      const __bf16* p = &sA[cur][(wm * 64 + mt * 16 + lr) * LROW + lh * 8];
      FragRaw r; r.lo = *(const v4u*)p; r.hi = *(const v4u*)(p + 16);
      afr[mt] = __builtin_bit_cast(v16bf, r);
    }
#pragma unroll
    for (int nt = 0; nt < 2; ++nt) {
      const __bf16* p = &sB[cur][(wn * 32 + nt * 16 + lr) * LROW + lh * 8];
      FragRaw r; r.lo = *(const v4u*)p; r.hi = *(const v4u*)(p + 16);
      bfr[nt] = __builtin_bit_cast(v16bf, r);
    }
#pragma unroll
    for (int mt = 0; mt < 4; ++mt)
#pragma unroll
      for (int nt = 0; nt < 2; ++nt)
        acc[mt][nt] = __builtin_amdgcn_wmma_f32_16x16x32_bf16(
            false, afr[mt], false, bfr[nt], (short)0, acc[mt][nt], false, false);

    if (more) storeA(cur ^ 1);
    wait_async0();
    __syncthreads();
  }

  // epilogue: C/D layout -> VGPR v holds row (v + lh*8), lane lr holds column
#pragma unroll
  for (int mt = 0; mt < 4; ++mt) {
    const int r0 = bm + wm * 64 + mt * 16 + lh * 8;
#pragma unroll
    for (int nt = 0; nt < 2; ++nt) {
      const int col = bn + wn * 32 + nt * 16 + lr;
      float bvv = 0.0f;
      if constexpr (HAS_BIAS) bvv = biasAll[col];
#pragma unroll
      for (int v = 0; v < 8; ++v) {
        const int row = r0 + v;
        float x = acc[mt][nt][v];
        if constexpr (HAS_BIAS) x += bvv;
        x *= scale;
        if constexpr (HAS_MASK) x += maskp[(long long)row * N + col];
        if constexpr (OUT_TRANS)      Cb[(long long)col * ldc + row] = (__bf16)x;
        else if constexpr (OUT_BF16)  Cb[(long long)row * ldc + col] = (__bf16)x;
        else                          Cf[(long long)row * ldc + col] = x;
      }
    }
  }
}

// ---------------- weight transpose + fp32 -> bf16 ----------------
__global__ __launch_bounds__(256)
void transpose_conv(const float* __restrict__ in, __bf16* __restrict__ out, int D)
{
  __shared__ float tile[32][33];
  const int x  = blockIdx.x * 32 + threadIdx.x;
  const int y0 = blockIdx.y * 32;
#pragma unroll
  for (int j = 0; j < 4; ++j)
    tile[threadIdx.y + j * 8][threadIdx.x] = in[(long long)(y0 + threadIdx.y + j * 8) * D + x];
  __syncthreads();
  const int x2 = blockIdx.y * 32 + threadIdx.x;
  const int y2 = blockIdx.x * 32;
#pragma unroll
  for (int j = 0; j < 4; ++j)
    out[(long long)(y2 + threadIdx.y + j * 8) * D + x2] = (__bf16)tile[threadIdx.x][threadIdx.y + j * 8];
}

// ---------------- row softmax: fp32 scores -> bf16 probs, aliased in place ----------------
// One block per row (S == 2048, 256 threads x 8 elems). bf16 result written over the
// row's own fp32 storage (all global reads happen before the first barrier -> safe).
__global__ __launch_bounds__(256)
void softmax_rows(float* __restrict__ scores)
{
  const long long row = blockIdx.x;
  float*  rp = scores + row * SEQ;
  __bf16* pp = (__bf16*)rp;
  const int tid = threadIdx.x;

  float v[8];
  float m = -3.402823466e38f;
#pragma unroll
  for (int i = 0; i < 8; ++i) { v[i] = rp[tid + i * 256]; m = fmaxf(m, v[i]); }

  __shared__ float red[256];
  red[tid] = m; __syncthreads();
  for (int s = 128; s > 0; s >>= 1) { if (tid < s) red[tid] = fmaxf(red[tid], red[tid + s]); __syncthreads(); }
  m = red[0]; __syncthreads();

  float sum = 0.0f;
#pragma unroll
  for (int i = 0; i < 8; ++i) { v[i] = __expf(v[i] - m); sum += v[i]; }
  red[tid] = sum; __syncthreads();
  for (int s = 128; s > 0; s >>= 1) { if (tid < s) red[tid] += red[tid + s]; __syncthreads(); }
  const float inv = 1.0f / red[0];

#pragma unroll
  for (int i = 0; i < 8; ++i) pp[tid + i * 256] = (__bf16)(v[i] * inv);
}

// ---------------- launcher ----------------
extern "C" void kernel_launch(void* const* d_in, const int* in_sizes, int n_in,
                              void* d_out, int out_size, void* d_ws, size_t ws_size,
                              hipStream_t stream)
{
  const float* Xq   = (const float*)d_in[0];
  const float* Xkv  = (const float*)d_in[1];
  const float* mask = (const float*)d_in[2];
  const float* Wq   = (const float*)d_in[3];
  const float* bq   = (const float*)d_in[4];
  const float* Wk   = (const float*)d_in[5];
  const float* bk   = (const float*)d_in[6];
  const float* Wv   = (const float*)d_in[7];
  const float* bv   = (const float*)d_in[8];
  float* out = (float*)d_out;

  char* ws = (char*)d_ws;
  size_t off = 0;
  auto carve = [&](size_t bytes) { void* p = ws + off; off += (bytes + 255) & ~(size_t)255; return p; };

  const size_t szW  = (size_t)DM * DM * sizeof(__bf16);          // 2 MB
  const size_t szSD = (size_t)NB * SEQ * DM * sizeof(__bf16);    // 16 MB
  const size_t szSc = (size_t)NB * SEQ * SEQ * sizeof(float);    // 64 MB

  __bf16* WqT    = (__bf16*)carve(szW);
  __bf16* WkT    = (__bf16*)carve(szW);
  __bf16* WvT    = (__bf16*)carve(szW);
  __bf16* Qbf    = (__bf16*)carve(szSD);   // pre-scaled by 1/sqrt(D_KEY)
  __bf16* Kbf    = (__bf16*)carve(szSD);
  __bf16* Vt     = (__bf16*)carve(szSD);   // per batch: [DM][SEQ]
  float*  scores = (float*) carve(szSc);   // also holds bf16 probs in place
  (void)ws_size; (void)in_sizes; (void)n_in; (void)out_size;

  // 1) weight transposes (fp32 -> bf16, W^T row-major N x K)
  {
    dim3 tb(32, 8), tg(DM / 32, DM / 32);
    transpose_conv<<<tg, tb, 0, stream>>>(Wq, WqT, DM);
    transpose_conv<<<tg, tb, 0, stream>>>(Wk, WkT, DM);
    transpose_conv<<<tg, tb, 0, stream>>>(Wv, WvT, DM);
  }

  const long long sSD = (long long)SEQ * DM;
  const long long sSS = (long long)SEQ * SEQ;

  // 2) projections (A fp32, output bf16)
  {
    dim3 g(DM / TN, SEQ / TM, NB);
    // Q' = (Xq Wq + bq) / 32
    gemm_wmma_bf16<true, false, true, false, true><<<g, 256, 0, stream>>>(
        Xq, WqT, bq, nullptr, Qbf, SEQ, DM, DM, DM, DM, DM,
        sSD, 0, sSD, 0, 0.03125f);
    // K = Xkv Wk + bk
    gemm_wmma_bf16<true, false, true, false, true><<<g, 256, 0, stream>>>(
        Xkv, WkT, bk, nullptr, Kbf, SEQ, DM, DM, DM, DM, DM,
        sSD, 0, sSD, 0, 1.0f);
    // V^T = (Xkv Wv + bv)^T   (ldc = SEQ, per-batch stride DM*SEQ)
    gemm_wmma_bf16<true, false, true, true, true><<<g, 256, 0, stream>>>(
        Xkv, WvT, bv, nullptr, Vt, SEQ, DM, DM, DM, DM, SEQ,
        sSD, 0, (long long)DM * SEQ, 0, 1.0f);
  }

  // 3) scores = Q' K^T + mask  (fp32 out)
  {
    dim3 g(SEQ / TN, SEQ / TM, NB);
    gemm_wmma_bf16<false, true, false, false, false><<<g, 256, 0, stream>>>(
        Qbf, Kbf, nullptr, mask, scores, SEQ, SEQ, DM, DM, DM, SEQ,
        sSD, sSD, sSS, sSS, 1.0f);
  }

  // 4) softmax rows -> bf16 probs written in place (row stride stays 4*SEQ bytes)
  softmax_rows<<<NB * SEQ, 256, 0, stream>>>(scores);

  // 5) O = P V  (A = bf16 probs with lda = 2*SEQ elems, BT = V^T), fp32 out
  {
    dim3 g(DM / TN, SEQ / TM, NB);
    gemm_wmma_bf16<false, false, false, false, false><<<g, 256, 0, stream>>>(
        (const __bf16*)scores, Vt, nullptr, nullptr, out, SEQ, DM, SEQ,
        2 * SEQ, SEQ, DM,
        2 * sSS, (long long)DM * SEQ, sSD, 0, 1.0f);
  }
}
